// SelfAttentionEncoder_13632226198097
// MI455X (gfx1250) — matrix operand order repaired and checked by reading, not compiled, not run on hardware
//
#include <hip/hip_runtime.h>
#include <hip/hip_bf16.h>

#define B_  4
#define S_  2048
#define D_  1024
#define H_  16
#define HD_ 64
#define BS_ (B_ * S_)   // 8192

typedef __attribute__((ext_vector_type(16))) _Float16 v16h;
typedef __attribute__((ext_vector_type(8)))  float    v8f;
typedef int vi4 __attribute__((vector_size(16)));  // matches builtin pointee

union Frag { uint4 u[2]; v16h h; };

#if __has_builtin(__builtin_amdgcn_global_load_async_to_lds_b128)
#define HAVE_ASYNC_LDS 1
#endif

// Copy 16B global -> LDS. Async (ASYNCcnt-tracked) when the toolchain has the
// gfx1250 builtin, otherwise a register-staged copy.
__device__ __forceinline__ void stage16(const _Float16* g, _Float16* l) {
#ifdef HAVE_ASYNC_LDS
  __builtin_amdgcn_global_load_async_to_lds_b128(
      (__attribute__((address_space(1))) vi4*)g,
      (__attribute__((address_space(3))) vi4*)l, 0, 0);
#else
  *(uint4*)(void*)l = *(const uint4*)(const void*)g;
#endif
}

__device__ __forceinline__ void wait_async() {
#ifdef HAVE_ASYNC_LDS
#if __has_builtin(__builtin_amdgcn_s_wait_asynccnt)
  __builtin_amdgcn_s_wait_asynccnt(0);
#else
  asm volatile("s_wait_asynccnt 0x0" ::: "memory");
#endif
#endif
}

// Load 2x16B fragments (A or B operand of v_wmma_f32_16x16x32_f16)
__device__ __forceinline__ v16h load_frag(const _Float16* p0, const _Float16* p1) {
  Frag f;
  f.u[0] = *(const uint4*)(const void*)p0;
  f.u[1] = *(const uint4*)(const void*)p1;
  return f.h;
}

__device__ __forceinline__ v8f wmma_f16(v16h a, v16h b, v8f c) {
  return __builtin_amdgcn_wmma_f32_16x16x32_f16(false, a, false, b, (short)0, c, false, false);
}

// ---------------------------------------------------------------- f32 -> f16
__global__ void cvt_f32_f16(const float* __restrict__ src,
                            _Float16* __restrict__ dst, int n) {
  int i = blockIdx.x * blockDim.x + threadIdx.x;
  int stride = gridDim.x * blockDim.x;
  for (; i < n; i += stride) dst[i] = (_Float16)src[i];
}

// ------------------------------------------------- Q/K/V projection (fused z)
// out[m,n] = sum_k x[m,k] * W[n,k].
// Register-blocked: one wave computes a 64x32 block = 4x2 grid of 16x16 tiles.
// z==0 -> Q [B,H,S,HD]; z==1 -> K [B,H,S,HD]; z==2 -> V transposed [B,H,HD,S].
__global__ __launch_bounds__(128) void qkv_proj(
    const _Float16* __restrict__ xh,
    const _Float16* __restrict__ Wqh,
    const _Float16* __restrict__ Wkh,
    const _Float16* __restrict__ Wvh,
    _Float16* __restrict__ Qh,
    _Float16* __restrict__ Kh,
    _Float16* __restrict__ Vt) {
  const int lane = threadIdx.x;
  const int r  = lane & 15;
  const int hf = lane >> 4;
  const int m0 = (blockIdx.x * blockDim.y + threadIdx.y) * 64;
  const int n0 = blockIdx.y * 32;
  const int z  = blockIdx.z;
  const _Float16* W = (z == 0) ? Wqh : (z == 1) ? Wkh : Wvh;

  const _Float16* a0 = xh + (size_t)(m0 + r) * D_;
  const _Float16* b0 = W  + (size_t)(n0 + r) * D_;

  v8f acc[4][2] = {};
#pragma unroll 2
  for (int k0 = 0; k0 < D_; k0 += 32) {
    __builtin_prefetch(a0 + k0 + 128, 0, 0);
    __builtin_prefetch(b0 + k0 + 128, 0, 0);
    v16h bfr[2];
#pragma unroll
    for (int j = 0; j < 2; ++j) {
      const _Float16* bp = b0 + (size_t)(16 * j) * D_ + k0 + 16 * hf;
      bfr[j] = load_frag(bp, bp + 8);
    }
#pragma unroll
    for (int i = 0; i < 4; ++i) {
      const _Float16* ap = a0 + (size_t)(16 * i) * D_ + k0;
      v16h afr = load_frag(ap + 16 * hf, ap + 16 * hf + 8);
#pragma unroll
      for (int j = 0; j < 2; ++j)
        acc[i][j] = wmma_f16(afr, bfr[j], acc[i][j]);
    }
  }
  // C layout per tile: lane holds N = n0+16j+r, rows M = m0+16i + v + 8*hf
  const int b = m0 >> 11;  // 64-row blocks never straddle a batch boundary
#pragma unroll
  for (int i = 0; i < 4; ++i) {
    const int sb = ((m0 + 16 * i) & 2047) + 8 * hf;
#pragma unroll
    for (int j = 0; j < 2; ++j) {
      const int d  = n0 + 16 * j + r;
      const int hh = d >> 6;
      const int hd = d & 63;
      if (z == 2) {
        // V transposed: Vt[(b,h), hd, s] — v rows contiguous in s
        _Float16* dst = Vt + ((size_t)(b * H_ + hh) * HD_ + hd) * S_ + sb;
#pragma unroll
        for (int v = 0; v < 8; ++v) dst[v] = (_Float16)acc[i][j][v];
      } else {
        _Float16* dst = ((z == 0) ? Qh : Kh) +
                        ((size_t)(b * H_ + hh) * S_ + sb) * HD_ + hd;
#pragma unroll
        for (int v = 0; v < 8; ++v) dst[(size_t)v * HD_] = (_Float16)acc[i][j][v];
      }
    }
  }
}

// ----------------------------------------------------- flash attention core
// Block = 4 waves, one (b,h), 64 query rows (16 per wave). Keys streamed in
// chunks of 32; each chunk's K (32x64) and V (64x32) staged once per block
// into LDS via async global->LDS, then consumed by all 4 waves via ds_load.
__global__ __launch_bounds__(128) void flash_attn(
    const _Float16* __restrict__ Qh,
    const _Float16* __restrict__ Kh,
    const _Float16* __restrict__ Vt,
    _Float16* __restrict__ Ah) {
  __shared__ __align__(16) _Float16 Kl[32][64];     // keys x hd   (4 KB)
  __shared__ __align__(16) _Float16 Vl[64][32];     // hd x keys   (4 KB)
  __shared__ __align__(16) _Float16 Pl[4][16][32];  // per-wave P  (4 KB)
  const int lane = threadIdx.x;
  const int r  = lane & 15;
  const int hf = lane >> 4;
  const int w  = threadIdx.y;
  const int tid = w * 32 + lane;  // 0..127
  const int bh = blockIdx.x;
  const int b  = bh >> 4;
  const int hh = bh & 15;
  const int q0 = (blockIdx.y * blockDim.y + w) * 16;

  const _Float16* Qb = Qh + (size_t)bh * S_ * HD_;
  const _Float16* Kb = Kh + (size_t)bh * S_ * HD_;
  const _Float16* Vb = Vt + (size_t)bh * HD_ * S_;

  // Q A-operands for Kdim 0..31 and 32..63, resident all loop
  const _Float16* qrow = Qb + (size_t)(q0 + r) * HD_;
  const v16h aQ0 = load_frag(qrow + 16 * hf,      qrow + 16 * hf + 8);
  const v16h aQ1 = load_frag(qrow + 32 + 16 * hf, qrow + 40 + 16 * hf);

  v8f acc[4] = {};
  float mrow[8], lrow[8];
#pragma unroll
  for (int v = 0; v < 8; ++v) { mrow[v] = -__builtin_inff(); lrow[v] = 0.f; }
  const float scale = 0.125f;  // 1/sqrt(HD)

  for (int kc = 0; kc < S_; kc += 32) {
    // ---- stage K chunk (contiguous 4KB) and V chunk (64 rows x 64B) ----
    {
      const _Float16* gk = Kb + (size_t)kc * HD_;
#pragma unroll
      for (int c = 0; c < 2; ++c) {
        const int idx = tid + 128 * c;                 // 0..255 16B chunks
        stage16(gk + idx * 8, &Kl[0][0] + idx * 8);
      }
#pragma unroll
      for (int c = 0; c < 2; ++c) {
        const int idx = tid + 128 * c;                 // row=idx/4, part=idx%4
        const int hd = idx >> 2, part = idx & 3;
        stage16(Vb + (size_t)hd * S_ + kc + part * 8, &Vl[0][0] + idx * 8);
      }
      wait_async();
    }
    __syncthreads();  // Kl/Vl valid for all waves

    // ---- scores: S[16q x 32k] from LDS K ----
    const _Float16* kr0 = &Kl[r][0];
    const _Float16* kr1 = &Kl[16 + r][0];
    v8f sc0 = {}, sc1 = {};
    {
      v16h bk = load_frag(kr0 + 16 * hf, kr0 + 16 * hf + 8);
      sc0 = wmma_f16(aQ0, bk, sc0);
      bk = load_frag(kr0 + 32 + 16 * hf, kr0 + 40 + 16 * hf);
      sc0 = wmma_f16(aQ1, bk, sc0);
      bk = load_frag(kr1 + 16 * hf, kr1 + 16 * hf + 8);
      sc1 = wmma_f16(aQ0, bk, sc1);
      bk = load_frag(kr1 + 32 + 16 * hf, kr1 + 40 + 16 * hf);
      sc1 = wmma_f16(aQ1, bk, sc1);
    }
    // ---- online softmax; row M=v+8*hf lives across 16 lanes of this half ----
#pragma unroll
    for (int v = 0; v < 8; ++v) {
      float x0 = sc0[v] * scale;
      float x1 = sc1[v] * scale;
      float cm = fmaxf(x0, x1);
#pragma unroll
      for (int off = 1; off < 16; off <<= 1) cm = fmaxf(cm, __shfl_xor(cm, off));
      float mn = fmaxf(mrow[v], cm);
      float p0 = __expf(x0 - mn);
      float p1 = __expf(x1 - mn);
      float fr = __expf(mrow[v] - mn);
      float rs = p0 + p1;
#pragma unroll
      for (int off = 1; off < 16; off <<= 1) rs += __shfl_xor(rs, off);
      lrow[v] = lrow[v] * fr + rs;
      mrow[v] = mn;
      acc[0][v] *= fr; acc[1][v] *= fr; acc[2][v] *= fr; acc[3][v] *= fr;
      Pl[w][v + 8 * hf][r]      = (_Float16)p0;   // wave-private slice;
      Pl[w][v + 8 * hf][16 + r] = (_Float16)p1;   // same-wave LDS is in-order
    }
    // ---- P: C-layout -> A-layout via LDS, then O += P @ V from LDS ----
    Frag fp;
    const uint4* prow = (const uint4*)(const void*)&Pl[w][r][0];
    fp.u[0] = prow[2 * hf];
    fp.u[1] = prow[2 * hf + 1];
    const v16h aP = fp.h;
#pragma unroll
    for (int t = 0; t < 4; ++t) {
      const _Float16* vr = &Vl[16 * t + r][16 * hf];
      v16h bv = load_frag(vr, vr + 8);
      acc[t] = wmma_f16(aP, bv, acc[t]);
    }
    __syncthreads();  // all consumers done before next chunk overwrites Kl/Vl
  }
  // normalize and store f16 [B,S,D] for the output projection
  _Float16* obase = Ah + ((size_t)b * S_ + q0 + 8 * hf) * D_ + hh * HD_ + r;
#pragma unroll
  for (int t = 0; t < 4; ++t) {
#pragma unroll
    for (int v = 0; v < 8; ++v)
      obase[(size_t)v * D_ + 16 * t] = (_Float16)(acc[t][v] / lrow[v]);
  }
}

// ------------------------------------------------------- output projection
// Same 64x32 register-blocked GEMM, f32 output.
__global__ __launch_bounds__(128) void out_proj(
    const _Float16* __restrict__ Ah,
    const _Float16* __restrict__ Woh,
    float* __restrict__ out) {
  const int lane = threadIdx.x;
  const int r  = lane & 15;
  const int hf = lane >> 4;
  const int m0 = (blockIdx.x * blockDim.y + threadIdx.y) * 64;
  const int n0 = blockIdx.y * 32;
  const _Float16* a0 = Ah  + (size_t)(m0 + r) * D_;
  const _Float16* b0 = Woh + (size_t)(n0 + r) * D_;

  v8f acc[4][2] = {};
#pragma unroll 2
  for (int k0 = 0; k0 < D_; k0 += 32) {
    __builtin_prefetch(a0 + k0 + 128, 0, 0);
    __builtin_prefetch(b0 + k0 + 128, 0, 0);
    v16h bfr[2];
#pragma unroll
    for (int j = 0; j < 2; ++j) {
      const _Float16* bp = b0 + (size_t)(16 * j) * D_ + k0 + 16 * hf;
      bfr[j] = load_frag(bp, bp + 8);
    }
#pragma unroll
    for (int i = 0; i < 4; ++i) {
      const _Float16* ap = a0 + (size_t)(16 * i) * D_ + k0;
      v16h afr = load_frag(ap + 16 * hf, ap + 16 * hf + 8);
#pragma unroll
      for (int j = 0; j < 2; ++j)
        acc[i][j] = wmma_f16(afr, bfr[j], acc[i][j]);
    }
  }
#pragma unroll
  for (int i = 0; i < 4; ++i) {
#pragma unroll
    for (int j = 0; j < 2; ++j) {
      float* dst = out + (size_t)(m0 + 16 * i + 8 * hf) * D_ + n0 + 16 * j + r;
#pragma unroll
      for (int v = 0; v < 8; ++v) dst[(size_t)v * D_] = acc[i][j][v];
    }
  }
}

// ---------------------------------------------------------------- launcher
extern "C" void kernel_launch(void* const* d_in, const int* in_sizes, int n_in,
                              void* d_out, int out_size, void* d_ws, size_t ws_size,
                              hipStream_t stream) {
  const float* x  = (const float*)d_in[0];
  const float* Wq = (const float*)d_in[1];
  const float* Wk = (const float*)d_in[2];
  const float* Wv = (const float*)d_in[3];
  const float* Wo = (const float*)d_in[4];
  float* out = (float*)d_out;

  char* ws = (char*)d_ws;
  size_t off = 0;
  auto alloc = [&](size_t bytes) -> void* {
    void* p = ws + off;
    off += (bytes + 255) & ~(size_t)255;
    return p;
  };
  _Float16* xh  = (_Float16*)alloc((size_t)BS_ * D_ * 2);
  _Float16* Wqh = (_Float16*)alloc((size_t)D_ * D_ * 2);
  _Float16* Wkh = (_Float16*)alloc((size_t)D_ * D_ * 2);
  _Float16* Wvh = (_Float16*)alloc((size_t)D_ * D_ * 2);
  _Float16* Woh = (_Float16*)alloc((size_t)D_ * D_ * 2);
  _Float16* Qh  = (_Float16*)alloc((size_t)BS_ * D_ * 2);
  _Float16* Kh  = (_Float16*)alloc((size_t)BS_ * D_ * 2);
  _Float16* Vt  = (_Float16*)alloc((size_t)BS_ * D_ * 2);
  _Float16* Ah  = (_Float16*)alloc((size_t)BS_ * D_ * 2);

  cvt_f32_f16<<<dim3(2048), dim3(256), 0, stream>>>(x,  xh,  BS_ * D_);
  cvt_f32_f16<<<dim3(512),  dim3(256), 0, stream>>>(Wq, Wqh, D_ * D_);
  cvt_f32_f16<<<dim3(512),  dim3(256), 0, stream>>>(Wk, Wkh, D_ * D_);
  cvt_f32_f16<<<dim3(512),  dim3(256), 0, stream>>>(Wv, Wvh, D_ * D_);
  cvt_f32_f16<<<dim3(512),  dim3(256), 0, stream>>>(Wo, Woh, D_ * D_);

  // 128 m-blocks(64) / 4 waves per block, 32 n-blocks(32), z = Q/K/V
  qkv_proj<<<dim3(BS_ / 256, D_ / 32, 3), dim3(32, 4), 0, stream>>>(
      xh, Wqh, Wkh, Wvh, Qh, Kh, Vt);

  // 64 (b,h) x 32 q-blocks (4 waves x 16 rows)
  flash_attn<<<dim3(B_ * H_, S_ / 64), dim3(32, 4), 0, stream>>>(Qh, Kh, Vt, Ah);

  out_proj<<<dim3(BS_ / 256, D_ / 32), dim3(32, 4), 0, stream>>>(Ah, Woh, out);
}
